// Net_37108517437447
// MI455X (gfx1250) — compile-verified, matrix operand order checked by reading, not run on hardware
//
#include <hip/hip_runtime.h>
#include <math.h>

// ---------------------------------------------------------------------------
// 2-layer GCN for MI455X (gfx1250, wave32).
//   h1 = relu( Dh^-1/2 (A+I) Dh^-1/2 (X @ W1) + b1 )
//   h2 =       Dh^-1/2 (A+I) Dh^-1/2 (h1 @ W2) + b2
//   out = log_softmax(h2)
// X@W1 is the only real GEMM (1.64 GFLOP, 205MB -> HBM bound) and is done
// with v_wmma_f32_16x16x32_bf16 (N=16 == one WMMA tile column, K=512 = 16
// k-steps). Everything per-edge (gather/scatter over 6.4MB tables) is
// L2-resident on the 192MB L2; fp32 scatter uses hardware global_atomic_add_f32.
// ---------------------------------------------------------------------------

typedef __attribute__((ext_vector_type(16))) __bf16 v16bf;
typedef __attribute__((ext_vector_type(8)))  __bf16 v8bf;
typedef __attribute__((ext_vector_type(8)))  float  v8f;
typedef __attribute__((ext_vector_type(4)))  float  v4f;

#define IN_FEAT 512
#define HID     16

__device__ __forceinline__ void atomAddF(float* p, float v) {
  // relaxed + agent scope -> global_atomic_add_f32 (no CAS loop)
  __hip_atomic_fetch_add(p, v, __ATOMIC_RELAXED, __HIP_MEMORY_SCOPE_AGENT);
}

// --------------------------- X @ W1 via WMMA -------------------------------
// One wave computes one 16x16 output tile. W1 staged in LDS transposed to
// [n][k] (K contiguous) so B fragments are two ds_load_b128 per k-step.
__global__ __launch_bounds__(256) void gemm1_wmma_k(
    const float* __restrict__ X, const float* __restrict__ W1,
    float* __restrict__ XW1, int nTiles) {
  __shared__ __bf16 Bs[HID * IN_FEAT];  // 16KB, [n][k]
  for (int i = threadIdx.x; i < HID * IN_FEAT; i += 256) {
    int k = i >> 4, n = i & 15;                 // W1 is [k][n], k*16+n
    Bs[n * IN_FEAT + k] = (__bf16)W1[i];
  }
  __syncthreads();

  int wave = threadIdx.x >> 5;
  int lane = threadIdx.x & 31;
  int tile = blockIdx.x * 8 + wave;
  if (tile >= nTiles) return;

  int rc   = lane & 15;     // A: row index; B/D: column index
  int half = lane >> 4;     // K-half select per ISA 16-bit layouts

  const float*  arow = X  + (size_t)(tile * 16 + rc) * IN_FEAT;
  const __bf16* bcol = Bs + (size_t)rc * IN_FEAT;

  v8f c = {0.f, 0.f, 0.f, 0.f, 0.f, 0.f, 0.f, 0.f};

#pragma unroll
  for (int ks = 0; ks < IN_FEAT / 32; ++ks) {
    int kb = ks * 32 + half * 8;
    // A fragment: elems 0..7 = K kb..kb+7, elems 8..15 = K kb+16..kb+23
    v4f a0 = *(const v4f*)(arow + kb);
    v4f a1 = *(const v4f*)(arow + kb + 4);
    v4f a2 = *(const v4f*)(arow + kb + 16);
    v4f a3 = *(const v4f*)(arow + kb + 20);
    v8bf b0 = *(const v8bf*)(bcol + kb);
    v8bf b1 = *(const v8bf*)(bcol + kb + 16);
    v16bf A, B;
#pragma unroll
    for (int i = 0; i < 4; ++i) {
      A[i]      = (__bf16)a0[i];
      A[4 + i]  = (__bf16)a1[i];
      A[8 + i]  = (__bf16)a2[i];
      A[12 + i] = (__bf16)a3[i];
    }
#pragma unroll
    for (int i = 0; i < 8; ++i) { B[i] = b0[i]; B[8 + i] = b1[i]; }
    c = __builtin_amdgcn_wmma_f32_16x16x32_bf16(
        /*neg_a=*/false, A, /*neg_b=*/false, B,
        /*c_mod=*/(short)0, c, /*reuse_a=*/false, /*reuse_b=*/false);
  }

  // D layout: lane holds column rc, rows half*8 + r in c[r]
  float* orow = XW1 + ((size_t)tile * 16 + half * 8) * HID + rc;
#pragma unroll
  for (int r = 0; r < 8; ++r) orow[r * HID] = c[r];
}

// --------------------------- degree / norm ---------------------------------
__global__ void deg_init_k(int* DEG, int N) {
  int n = blockIdx.x * blockDim.x + threadIdx.x;
  if (n < N) DEG[n] = 1;  // self loop
}
__global__ void deg_edge_k(const int* __restrict__ DST, int* DEG, int E) {
  int e = blockIdx.x * blockDim.x + threadIdx.x;
  if (e < E) atomicAdd(&DEG[DST[e]], 1);
}
__global__ void dinv_k(float* DNV, int N) {
  int n = blockIdx.x * blockDim.x + threadIdx.x;
  if (n < N) {
    int c = ((const int*)DNV)[n];   // deg >= 1 always (self loop)
    DNV[n] = rsqrtf((float)c);
  }
}

// ---------------- layer-1 aggregation: self term, scatter, relu ------------
__global__ void self1_k(const float* __restrict__ XW1,
                        const float* __restrict__ DNV, float* H, int N) {
  int t = blockIdx.x * blockDim.x + threadIdx.x;
  int n = t >> 2;
  if (n >= N) return;
  int q = t & 3;
  float w = DNV[n] * DNV[n];
  v4f v = *(const v4f*)(XW1 + (size_t)n * HID + q * 4);
  v4f o;
#pragma unroll
  for (int i = 0; i < 4; ++i) o[i] = w * v[i];
  *(v4f*)(H + (size_t)n * HID + q * 4) = o;
}

__global__ void agg1_k(const int* __restrict__ SRC, const int* __restrict__ DST,
                       const float* __restrict__ DNV,
                       const float* __restrict__ XW1, float* H, int E) {
  int t = blockIdx.x * blockDim.x + threadIdx.x;
  int e = t >> 2;
  if (e >= E) return;
  int q = t & 3;
  int s = SRC[e], d = DST[e];
  float nrm = DNV[s] * DNV[d];
  v4f v = *(const v4f*)(XW1 + (size_t)s * HID + q * 4);
  float* hp = H + (size_t)d * HID + q * 4;
#pragma unroll
  for (int i = 0; i < 4; ++i) atomAddF(hp + i, nrm * v[i]);
}

__global__ void relu_bias_k(float* H, const float* __restrict__ B1, int N) {
  int t = blockIdx.x * blockDim.x + threadIdx.x;
  int n = t >> 2;
  if (n >= N) return;
  int q = t & 3;
  v4f b = *(const v4f*)(B1 + q * 4);
  float* hp = H + (size_t)n * HID + q * 4;
  v4f v = *(const v4f*)hp;
#pragma unroll
  for (int i = 0; i < 4; ++i) v[i] = fmaxf(v[i] + b[i], 0.f);
  *(v4f*)hp = v;
}

// ---------------- layer 2: tiny GEMM + self term, scatter, logsoftmax ------
__global__ void gemm2_k(const float* __restrict__ H, const float* __restrict__ W2,
                        const float* __restrict__ DNV, float* HW2, float* OUT2,
                        int N) {
  int n = blockIdx.x * blockDim.x + threadIdx.x;
  if (n >= N) return;
  const float* hr = H + (size_t)n * HID;
  float h0 = 0.f, h1 = 0.f;
#pragma unroll
  for (int f = 0; f < HID; ++f) {
    float hv = hr[f];
    h0 += hv * W2[2 * f];
    h1 += hv * W2[2 * f + 1];
  }
  HW2[2 * n] = h0;
  HW2[2 * n + 1] = h1;
  float w = DNV[n] * DNV[n];
  OUT2[2 * n] = w * h0;
  OUT2[2 * n + 1] = w * h1;
}

__global__ void agg2_k(const int* __restrict__ SRC, const int* __restrict__ DST,
                       const float* __restrict__ DNV,
                       const float* __restrict__ HW2, float* OUT2, int E) {
  int e = blockIdx.x * blockDim.x + threadIdx.x;
  if (e >= E) return;
  int s = SRC[e], d = DST[e];
  float nrm = DNV[s] * DNV[d];
  atomAddF(&OUT2[2 * d],     nrm * HW2[2 * s]);
  atomAddF(&OUT2[2 * d + 1], nrm * HW2[2 * s + 1]);
}

__global__ void lsm_k(const float* __restrict__ OUT2, const float* __restrict__ B2,
                      float* __restrict__ out, int N) {
  int n = blockIdx.x * blockDim.x + threadIdx.x;
  if (n >= N) return;
  float a = OUT2[2 * n]     + B2[0];
  float b = OUT2[2 * n + 1] + B2[1];
  float m = fmaxf(a, b);
  float lse = m + logf(expf(a - m) + expf(b - m));
  out[2 * n]     = a - lse;
  out[2 * n + 1] = b - lse;
}

// ---------------------------------------------------------------------------
extern "C" void kernel_launch(void* const* d_in, const int* in_sizes, int n_in,
                              void* d_out, int out_size, void* d_ws, size_t ws_size,
                              hipStream_t stream) {
  const float* X  = (const float*)d_in[0];
  const int*   EI = (const int*)d_in[1];   // [2, E] flat, int32 per harness
  const float* W1 = (const float*)d_in[2];
  const float* B1 = (const float*)d_in[3];
  const float* W2 = (const float*)d_in[4];
  const float* B2 = (const float*)d_in[5];

  const int N = in_sizes[0] / IN_FEAT;     // 100000
  const int E = in_sizes[1] / 2;           // 3200000
  const int* SRC = EI;
  const int* DST = EI + E;

  // workspace layout (floats): XW1 | H | DINV(deg) | HW2 | OUT2  (~14.8MB)
  float* ws   = (float*)d_ws;
  float* XW1  = ws;
  float* H    = ws + (size_t)N * HID;
  float* DNV  = ws + (size_t)2 * N * HID;
  float* HW2  = DNV + N;
  float* OUT2 = HW2 + (size_t)2 * N;

  const int thr = 256;
  const int nTiles = (N + 15) / 16;        // 6250, exact

  gemm1_wmma_k<<<(nTiles + 7) / 8, thr, 0, stream>>>(X, W1, XW1, nTiles);
  deg_init_k<<<(N + thr - 1) / thr, thr, 0, stream>>>((int*)DNV, N);
  deg_edge_k<<<(E + thr - 1) / thr, thr, 0, stream>>>(DST, (int*)DNV, E);
  dinv_k<<<(N + thr - 1) / thr, thr, 0, stream>>>(DNV, N);
  self1_k<<<(4 * N + thr - 1) / thr, thr, 0, stream>>>(XW1, DNV, H, N);
  agg1_k<<<(4 * E + thr - 1) / thr, thr, 0, stream>>>(SRC, DST, DNV, XW1, H, E);
  relu_bias_k<<<(4 * N + thr - 1) / thr, thr, 0, stream>>>(H, B1, N);
  gemm2_k<<<(N + thr - 1) / thr, thr, 0, stream>>>(H, W2, DNV, HW2, OUT2, N);
  agg2_k<<<(E + thr - 1) / thr, thr, 0, stream>>>(SRC, DST, DNV, HW2, OUT2, E);
  lsm_k<<<(N + thr - 1) / thr, thr, 0, stream>>>(OUT2, B2, (float*)d_out, N);
}
